// AttentionBlock_16398185136286
// MI455X (gfx1250) — matrix operand order repaired and checked by reading, not compiled, not run on hardware
//
#include <hip/hip_runtime.h>
#include <hip/hip_bf16.h>

// AttentionBlock: B=4, C=256, L=4096, GROUPS=8, f32 in/out.
// Pipeline: wconvert(f16 weights) + gn_stats -> gn_apply (LDS tile transpose,
//           async LDS->global store) -> qkv WMMA GEMM (pipelined)
//           -> fused 16-query-block attention (256KB LDS score strip,
//              TDM-staged K tiles, async-staged V tiles)
//           -> out-proj WMMA GEMM + residual.
// Matrix ops: v_wmma_f32_16x16x32_f16 (wave32, CDNA5).

typedef __attribute__((ext_vector_type(16))) _Float16     v16h;
typedef __attribute__((ext_vector_type(8)))  _Float16     v8h;
typedef __attribute__((ext_vector_type(8)))  float        v8f;
typedef __attribute__((ext_vector_type(4)))  unsigned int u32x4;
typedef __attribute__((ext_vector_type(8)))  unsigned int u32x8;

constexpr int   kB   = 4;
constexpr int   kC   = 256;
constexpr int   kL   = 4096;
constexpr int   kG   = 8;
constexpr int   kCG  = kC / kG;          // 32 channels per group
constexpr float kEps = 1e-5f;
constexpr float kScale = 0.0625f;        // 256^-0.5

static __device__ __forceinline__ v8f wmma_f16(v16h a, v16h b, v8f c) {
  return __builtin_amdgcn_wmma_f32_16x16x32_f16(false, a, false, b, (short)0, c,
                                                false, false);
}

// K index of VGPR pair v for A-fragment lanes (contiguous order: k = 16*half + 2v).
static __device__ __forceinline__ int a_kbase(int v, int half) {
  return 2 * v + 16 * half;
}

// A fragment (16x32 f16): the per-lane K pattern is one contiguous 16-half
// run (k = 16*half + s), so two 16B vector loads fill the whole fragment.
static __device__ __forceinline__ v16h load_a_f16(const _Float16* __restrict__ src,
                                                  int ld, int half, int m) {
  v8h lo = *(const v8h*)(src + m * ld + 16 * half);
  v8h hi = *(const v8h*)(src + m * ld + 16 * half + 8);
  v16h a;
#pragma unroll
  for (int i = 0; i < 8; ++i) {
    a[i] = lo[i];
    a[8 + i] = hi[i];
  }
  return a;
}

// ---- CDNA5 async / tensor data movement helpers ---------------------------

// Per-lane 16B global->LDS async copy (ASYNCcnt).
static __device__ __forceinline__ void async_copy_b128(const void* g, void* lds) {
  unsigned int l = (unsigned int)(unsigned long long)lds;
  asm volatile("global_load_async_to_lds_b128 %0, %1, off"
               :: "v"(l), "v"(g) : "memory");
}
// Per-lane 16B LDS->global async store (ASYNCcnt).
static __device__ __forceinline__ void async_store_b128(void* g, const void* lds) {
  unsigned int l = (unsigned int)(unsigned long long)lds;
  asm volatile("global_store_async_from_lds_b128 %0, %1, off"
               :: "v"(g), "v"(l) : "memory");
}
#define ASYNC_WAIT_0() asm volatile("s_wait_asynccnt 0x0" ::: "memory")
#define ASYNC_WAIT_8() asm volatile("s_wait_asynccnt 0x8" ::: "memory")

// TDM: DMA a 2-D tile (tile_d0 x tile_d1 elements, 2-byte data) from a
// row-major tensor (row length = stride0 elements) into LDS.
// D# per CDNA5 ISA 8.3/8.4.
static __device__ __forceinline__ void tdm_load_2d_f16(unsigned int lds_addr,
                                                       const void* gaddr,
                                                       unsigned int tile_d0,
                                                       unsigned int tile_d1,
                                                       unsigned int tens_d0,
                                                       unsigned int tens_d1,
                                                       unsigned int stride0) {
  unsigned long long ga = (unsigned long long)gaddr;
  u32x4 g0;
  g0[0] = 1u;                                        // count=1, user mode
  g0[1] = lds_addr;
  g0[2] = (unsigned int)ga;
  g0[3] = (unsigned int)((ga >> 32) & 0x01FFFFFFu) | (2u << 30);  // type=2
  unsigned long long s1 = (unsigned long long)stride0 * tens_d1;
  u32x8 g1;
  g1[0] = 1u << 16;                                  // data_size = 2 bytes
  g1[1] = (tens_d0 & 0xFFFFu) << 16;
  g1[2] = ((tens_d0 >> 16) & 0xFFFFu) | ((tens_d1 & 0xFFFFu) << 16);
  g1[3] = ((tens_d1 >> 16) & 0xFFFFu) | (tile_d0 << 16);
  g1[4] = tile_d1;
  g1[5] = stride0;
  g1[6] = ((unsigned int)s1 & 0xFFFFu) << 16;
  g1[7] = (unsigned int)(s1 >> 16);
  asm volatile("tensor_load_to_lds %0, %1" :: "s"(g0), "s"(g1) : "memory");
}

// ---------------------------------------------------------------------------
// Kernel 0: one-shot f32 -> f16 conversion of qkv_w (768x256) and
// out_w (256x256) so GEMM A-fragments are pure 16B f16 loads.
// ---------------------------------------------------------------------------
__global__ void wconvert_kernel(const float* __restrict__ qkv_w,
                                const float* __restrict__ out_w,
                                _Float16* __restrict__ qkv_wh,
                                _Float16* __restrict__ out_wh) {
  const int idx = blockIdx.x * 256 + threadIdx.x;  // 262144 total
  if (idx < 768 * kC) {
    qkv_wh[idx] = (_Float16)qkv_w[idx];
  } else {
    const int j = idx - 768 * kC;
    out_wh[j] = (_Float16)out_w[j];
  }
}

// ---------------------------------------------------------------------------
// Kernel 1: GroupNorm statistics. One block per (b, g).
// ---------------------------------------------------------------------------
__global__ void gn_stats_kernel(const float* __restrict__ x,
                                float* __restrict__ stats) {
  __shared__ float ssum[256];
  __shared__ float ssq[256];
  const int bg = blockIdx.x;
  const int b = bg / kG, g = bg % kG;
  const int tid = threadIdx.x;
  const int cg0 = g * kCG;
  const float* xb = x + (size_t)b * kC * kL;

  float s = 0.f, sq = 0.f;
  for (int i = tid; i < kCG * kL; i += 256) {
    int c = cg0 + (i >> 12);
    int l = i & (kL - 1);
    float v = xb[c * kL + l];
    s += v;
    sq += v * v;
  }
  ssum[tid] = s;
  ssq[tid] = sq;
  __syncthreads();
  for (int off = 128; off > 0; off >>= 1) {
    if (tid < off) {
      ssum[tid] += ssum[tid + off];
      ssq[tid] += ssq[tid + off];
    }
    __syncthreads();
  }
  if (tid == 0) {
    const float inv_n = 1.0f / (float)(kCG * kL);
    float mean = ssum[0] * inv_n;
    float var = ssq[0] * inv_n - mean * mean;
    stats[2 * bg]     = mean;
    stats[2 * bg + 1] = rsqrtf(var + kEps);
  }
}

// ---------------------------------------------------------------------------
// Kernel 2: apply GroupNorm + transpose to hT[b][l][c] (f16) via a 64x64
// LDS tile: coalesced f32 reads, coalesced async LDS->global f16 stores.
// ---------------------------------------------------------------------------
__global__ void gn_apply_kernel(const float* __restrict__ x,
                                const float* __restrict__ gn_w,
                                const float* __restrict__ gn_b,
                                const float* __restrict__ stats,
                                _Float16* __restrict__ hT) {
  __shared__ _Float16 tile[64 * 72];     // pad 72 to dodge bank conflicts
  const int tid = threadIdx.x;
  const int blk = blockIdx.x;            // b*256 + ct*64 + lt
  const int lt = blk & 63;
  const int ct = (blk >> 6) & 3;
  const int b = blk >> 8;
  const int c0 = ct * 64, l0 = lt * 64;

  const int cr = tid >> 6;               // 0..3
  const int ll = tid & 63;
#pragma unroll
  for (int i = 0; i < 16; ++i) {
    const int cl = i * 4 + cr;
    const int c = c0 + cl;
    const int bg = b * kG + (c >> 5);
    const float mean = stats[2 * bg];
    const float rstd = stats[2 * bg + 1];
    const float h =
        (x[(size_t)b * kC * kL + (size_t)c * kL + l0 + ll] - mean) * rstd *
            gn_w[c] + gn_b[c];
    tile[ll * 72 + cl] = (_Float16)h;
  }
  __syncthreads();

  // Write out 64 rows x 128B, 32B per thread as 2 async b128 stores.
  const int lrow = tid >> 2;             // 0..63
  const int chunk = tid & 3;             // 16-half chunks
  _Float16* dst = hT + (size_t)b * kL * kC + (size_t)(l0 + lrow) * kC + c0 +
                  chunk * 16;
  const _Float16* srcl = tile + lrow * 72 + chunk * 16;
  async_store_b128(dst, srcl);
  async_store_b128(dst + 8, srcl + 8);
  ASYNC_WAIT_0();
}

// ---------------------------------------------------------------------------
// Kernel 3: QKV GEMM (f16 weights). M=768, N=4096, K=256 per batch; each
// wave does a 16x64 strip with a 2-stage software pipeline.
// ---------------------------------------------------------------------------
__global__ void qkv_gemm_kernel(const _Float16* __restrict__ qkv_wh,
                                const float* __restrict__ qkv_b,
                                const _Float16* __restrict__ hT,
                                _Float16* __restrict__ qT,
                                _Float16* __restrict__ kT,
                                _Float16* __restrict__ vv) {
  const int lane = threadIdx.x & 31;
  const int half = lane >> 4;
  const int m = lane & 15;
  const int gwave = blockIdx.x * 8 + (threadIdx.x >> 5);  // 12288 waves
  const int nc = gwave & 63;
  const int mt = (gwave >> 6) % 48;
  const int b = gwave / (64 * 48);
  const int o0 = mt * 16;
  const int n0 = nc * 64;
  const _Float16* hb = hT + (size_t)b * kL * kC;
  const _Float16* wbase = qkv_wh + o0 * kC;

  __builtin_prefetch(wbase + m * kC, 0, 3);

  v8f acc0 = {}, acc1 = {}, acc2 = {}, acc3 = {};
  v16h a = load_a_f16(wbase, kC, half, m);
  const _Float16* bp0 = hb + 16 * half;
  v16h b0 = *(const v16h*)(bp0 + (size_t)(n0 + 0 * 16 + m) * kC);
  v16h b1 = *(const v16h*)(bp0 + (size_t)(n0 + 1 * 16 + m) * kC);
  v16h b2 = *(const v16h*)(bp0 + (size_t)(n0 + 2 * 16 + m) * kC);
  v16h b3 = *(const v16h*)(bp0 + (size_t)(n0 + 3 * 16 + m) * kC);
#pragma unroll
  for (int k0 = 0; k0 < kC; k0 += 32) {
    v16h an = a, n0b = b0, n1b = b1, n2b = b2, n3b = b3;
    if (k0 + 32 < kC) {                  // prefetch next stage
      an = load_a_f16(wbase + k0 + 32, kC, half, m);
      const _Float16* bp = hb + k0 + 32 + 16 * half;
      n0b = *(const v16h*)(bp + (size_t)(n0 + 0 * 16 + m) * kC);
      n1b = *(const v16h*)(bp + (size_t)(n0 + 1 * 16 + m) * kC);
      n2b = *(const v16h*)(bp + (size_t)(n0 + 2 * 16 + m) * kC);
      n3b = *(const v16h*)(bp + (size_t)(n0 + 3 * 16 + m) * kC);
    }
    acc0 = wmma_f16(a, b0, acc0);
    acc1 = wmma_f16(a, b1, acc1);
    acc2 = wmma_f16(a, b2, acc2);
    acc3 = wmma_f16(a, b3, acc3);
    a = an; b0 = n0b; b1 = n1b; b2 = n2b; b3 = n3b;
  }

  v8f accs[4] = {acc0, acc1, acc2, acc3};
#pragma unroll
  for (int j = 0; j < 4; ++j) {
    const int n = n0 + j * 16 + m;
#pragma unroll
    for (int r = 0; r < 8; ++r) {
      const int o = o0 + r + 8 * half;
      const _Float16 val = (_Float16)(accs[j][r] + qkv_b[o]);
      if (o < 256)
        qT[(size_t)b * kL * kC + (size_t)n * kC + o] = val;
      else if (o < 512)
        kT[(size_t)b * kL * kC + (size_t)n * kC + (o - 256)] = val;
      else
        vv[(size_t)b * kC * kL + (size_t)(o - 512) * kL + n] = val;
    }
  }
}

// ---------------------------------------------------------------------------
// Kernel 4: fused attention, one 16-query block per workgroup.
// 128 threads = 4 waves; 320KB LDS (256KB score strip + 64KB staging).
// Pass 1: TDM-staged K tiles (TENSORcnt). Pass 2: async-staged V tiles
// (ASYNCcnt), P fragments rebuilt from LDS with exp folded into the load.
// ---------------------------------------------------------------------------
__global__ void attn_kernel(const _Float16* __restrict__ qT,
                            const _Float16* __restrict__ kT,
                            const _Float16* __restrict__ vv,
                            _Float16* __restrict__ oT) {
  extern __shared__ char smem_raw[];
  float* S = (float*)smem_raw;                   // [16][4096] f32
  char* stage = (char*)(S + 16 * kL);            // 64KB staging region
  float* red       = (float*)stage;              // stats overlay
  float* rowmax    = red + 128;
  float* rowsuminv = rowmax + 16;

  const int tid = threadIdx.x;
  const int w = tid >> 5;
  const int lane = tid & 31;
  const int half = lane >> 4;
  const int m = lane & 15;
  const int b = blockIdx.x >> 8;
  const int q0 = (blockIdx.x & 255) * 16;

  const _Float16* qb = qT + (size_t)b * kL * kC;
  const _Float16* kb = kT + (size_t)b * kL * kC;
  const _Float16* vb = vv + (size_t)b * kC * kL;

  char* buf0 = stage + w * 16384;
  char* buf1 = buf0 + 8192;

  v16h aq[8];
#pragma unroll
  for (int ck = 0; ck < 8; ++ck)
    aq[ck] = load_a_f16(qb + (size_t)q0 * kC + ck * 32, kC, half, m);

  // ---- Pass 1: S = scale * Q K^T; K tiles DMA'd by the TDM.
  {
    const int kt0 = w * 64;
    tdm_load_2d_f16((unsigned int)(unsigned long long)buf0,
                    kb + (size_t)(kt0 * 16) * kC, kC, 16, kC, kL, kC);
    for (int i = 0; i < 64; ++i) {
      const int key0 = (kt0 + i) * 16;
      char* cur = (i & 1) ? buf1 : buf0;
      if (i + 1 < 64) {
        char* nxt = (i & 1) ? buf0 : buf1;
        tdm_load_2d_f16((unsigned int)(unsigned long long)nxt,
                        kb + (size_t)(key0 + 16) * kC, kC, 16, kC, kL, kC);
        __builtin_amdgcn_s_wait_tensorcnt((short)1);
      } else {
        __builtin_amdgcn_s_wait_tensorcnt((short)0);
      }
      v8f acc = {};
#pragma unroll
      for (int ck = 0; ck < 8; ++ck) {
        v16h bf = *(const v16h*)(cur + m * 512 + ck * 64 + half * 32);
        acc = wmma_f16(aq[ck], bf, acc);
      }
#pragma unroll
      for (int r = 0; r < 8; ++r)
        S[(r + 8 * half) * kL + key0 + m] = acc[r] * kScale;
    }
  }
  __syncthreads();

  // ---- Block softmax stats: 8 threads per row of the 16x4096 strip.
  {
    const int row = tid >> 3;
    const int sub = tid & 7;
    const float* srow = S + row * kL;
    float mx = -3.4e38f;
    for (int i = sub * 512; i < sub * 512 + 512; ++i) mx = fmaxf(mx, srow[i]);
    red[tid] = mx;
    __syncthreads();
    if (sub == 0) {
      float rm = red[tid];
#pragma unroll
      for (int j = 1; j < 8; ++j) rm = fmaxf(rm, red[tid + j]);
      rowmax[row] = rm;
    }
    __syncthreads();
    const float rm = rowmax[row];
    float sum = 0.f;
    for (int i = sub * 512; i < sub * 512 + 512; ++i)
      sum += __expf(srow[i] - rm);
    red[tid] = sum;
    __syncthreads();
    if (sub == 0) {
      float rs = red[tid];
#pragma unroll
      for (int j = 1; j < 8; ++j) rs += red[tid + j];
      rowsuminv[row] = 1.0f / rs;
    }
    __syncthreads();
  }

  const float mrow = rowmax[m];
  const float sinv = rowsuminv[m];
  __syncthreads();                       // stats snapshotted; free staging

  // ---- Pass 2: O(16 x 64 d-slice) = P x V^T, async-staged V tiles.
  const int cbase = w * 64;
  auto stage_v = [&](char* dst, int kk0) {
#pragma unroll
    for (int i = 0; i < 8; ++i) {
      const int row = i * 8 + (lane >> 2);
      const int col = (lane & 3) * 8;
      async_copy_b128(vb + (size_t)(cbase + row) * kL + kk0 + col,
                      dst + row * 64 + col * 2);
    }
  };

  v8f o0a = {}, o1a = {}, o2a = {}, o3a = {};
  stage_v(buf0, 0);
  for (int kt2 = 0; kt2 < 128; ++kt2) {
    const int kk0 = kt2 * 32;
    char* cur = (kt2 & 1) ? buf1 : buf0;
    if (kt2 + 1 < 128) {
      stage_v((kt2 & 1) ? buf0 : buf1, kk0 + 32);
      ASYNC_WAIT_8();
    } else {
      ASYNC_WAIT_0();
    }
    v16h ap;
#pragma unroll
    for (int v = 0; v < 8; ++v) {
      const int kb2 = a_kbase(v, half);
      const float s0 = S[m * kL + kk0 + kb2];
      const float s1 = S[m * kL + kk0 + kb2 + 1];
      ap[2 * v]     = (_Float16)(__expf(s0 - mrow) * sinv);
      ap[2 * v + 1] = (_Float16)(__expf(s1 - mrow) * sinv);
    }
    v16h b0 = *(const v16h*)(cur + (0 * 16 + m) * 64 + half * 32);
    v16h b1 = *(const v16h*)(cur + (1 * 16 + m) * 64 + half * 32);
    v16h b2 = *(const v16h*)(cur + (2 * 16 + m) * 64 + half * 32);
    v16h b3 = *(const v16h*)(cur + (3 * 16 + m) * 64 + half * 32);
    o0a = wmma_f16(ap, b0, o0a);
    o1a = wmma_f16(ap, b1, o1a);
    o2a = wmma_f16(ap, b2, o2a);
    o3a = wmma_f16(ap, b3, o3a);
  }

  v8f oacc[4] = {o0a, o1a, o2a, o3a};
#pragma unroll
  for (int j = 0; j < 4; ++j) {
#pragma unroll
    for (int r = 0; r < 8; ++r) {
      const int l = q0 + r + 8 * half;
      oT[(size_t)b * kL * kC + (size_t)l * kC + cbase + j * 16 + m] =
          (_Float16)oacc[j][r];
    }
  }
}

// ---------------------------------------------------------------------------
// Kernel 5: out-projection GEMM (f16 weights) + bias + residual, f32 out.
// ---------------------------------------------------------------------------
__global__ void proj_kernel(const _Float16* __restrict__ out_wh,
                            const float* __restrict__ out_b,
                            const _Float16* __restrict__ oT,
                            const float* __restrict__ x,
                            float* __restrict__ out) {
  const int lane = threadIdx.x & 31;
  const int half = lane >> 4;
  const int m = lane & 15;
  const int gwave = blockIdx.x * 8 + (threadIdx.x >> 5);  // 4096 waves
  const int nc = gwave & 63;
  const int mt = (gwave >> 6) & 15;
  const int b = gwave >> 10;
  const int o0 = mt * 16;
  const int n0 = nc * 64;
  const _Float16* ob = oT + (size_t)b * kL * kC;
  const _Float16* wbase = out_wh + o0 * kC;

  __builtin_prefetch(wbase + m * kC, 0, 3);

  v8f acc0 = {}, acc1 = {}, acc2 = {}, acc3 = {};
  v16h a = load_a_f16(wbase, kC, half, m);
  const _Float16* bp0 = ob + 16 * half;
  v16h b0 = *(const v16h*)(bp0 + (size_t)(n0 + 0 * 16 + m) * kC);
  v16h b1 = *(const v16h*)(bp0 + (size_t)(n0 + 1 * 16 + m) * kC);
  v16h b2 = *(const v16h*)(bp0 + (size_t)(n0 + 2 * 16 + m) * kC);
  v16h b3 = *(const v16h*)(bp0 + (size_t)(n0 + 3 * 16 + m) * kC);
#pragma unroll
  for (int k0 = 0; k0 < kC; k0 += 32) {
    v16h an = a, n0b = b0, n1b = b1, n2b = b2, n3b = b3;
    if (k0 + 32 < kC) {
      an = load_a_f16(wbase + k0 + 32, kC, half, m);
      const _Float16* bp = ob + k0 + 32 + 16 * half;
      n0b = *(const v16h*)(bp + (size_t)(n0 + 0 * 16 + m) * kC);
      n1b = *(const v16h*)(bp + (size_t)(n0 + 1 * 16 + m) * kC);
      n2b = *(const v16h*)(bp + (size_t)(n0 + 2 * 16 + m) * kC);
      n3b = *(const v16h*)(bp + (size_t)(n0 + 3 * 16 + m) * kC);
    }
    acc0 = wmma_f16(a, b0, acc0);
    acc1 = wmma_f16(a, b1, acc1);
    acc2 = wmma_f16(a, b2, acc2);
    acc3 = wmma_f16(a, b3, acc3);
    a = an; b0 = n0b; b1 = n1b; b2 = n2b; b3 = n3b;
  }

  v8f accs[4] = {acc0, acc1, acc2, acc3};
#pragma unroll
  for (int j = 0; j < 4; ++j) {
    const int n = n0 + j * 16 + m;
#pragma unroll
    for (int r = 0; r < 8; ++r) {
      const int o = o0 + r + 8 * half;
      const size_t idx = (size_t)b * kC * kL + (size_t)o * kL + n;
      out[idx] = x[idx] + out_b[o] + accs[j][r];
    }
  }
}

// ---------------------------------------------------------------------------
extern "C" void kernel_launch(void* const* d_in, const int* in_sizes, int n_in,
                              void* d_out, int out_size, void* d_ws, size_t ws_size,
                              hipStream_t stream) {
  const float* x     = (const float*)d_in[0];
  const float* gn_w  = (const float*)d_in[1];
  const float* gn_b  = (const float*)d_in[2];
  const float* qkv_w = (const float*)d_in[3];
  const float* qkv_b = (const float*)d_in[4];
  const float* out_w = (const float*)d_in[5];
  const float* out_b = (const float*)d_in[6];
  float* out = (float*)d_out;

  // Workspace (~42.5 MB): stats + hT/qT/kT/v/oT (5 x 8MB f16) + f16 weights.
  const size_t plane = (size_t)kB * kC * kL;   // 4194304 elements
  float*    stats = (float*)d_ws;
  _Float16* hT = (_Float16*)((char*)d_ws + 1024);
  _Float16* qT = hT + plane;
  _Float16* kT = qT + plane;
  _Float16* vv = kT + plane;
  _Float16* oT = vv + plane;
  _Float16* qkv_wh = oT + plane;               // 768*256
  _Float16* out_wh = qkv_wh + 768 * kC;        // 256*256

  wconvert_kernel<<<dim3(1024), dim3(256), 0, stream>>>(qkv_w, out_w,
                                                        qkv_wh, out_wh);

  gn_stats_kernel<<<dim3(kB * kG), dim3(256), 0, stream>>>(x, stats);

  gn_apply_kernel<<<dim3(kB * 4 * 64), dim3(256), 0, stream>>>(
      x, gn_w, gn_b, stats, hT);

  qkv_gemm_kernel<<<dim3(12288 / 8), dim3(256), 0, stream>>>(
      qkv_wh, qkv_b, hT, qT, kT, vv);

  const int flash_lds = 16 * kL * 4 + 64 * 1024;   // 320KB WGP max
  attn_kernel<<<dim3(kB * (kL / 16)), dim3(128), flash_lds, stream>>>(
      qT, kT, vv, oT);

  proj_kernel<<<dim3(4096 / 8), dim3(256), 0, stream>>>(
      out_wh, out_b, oT, x, out);
}
